// ActorCriticGAT_V2_56264071577780
// MI455X (gfx1250) — matrix operand order repaired and checked by reading, not compiled, not run on hardware
//
#include <hip/hip_runtime.h>
#include <hip/hip_bf16.h>

// ---------------------------------------------------------------------------
// ActorCriticGAT on MI455X (gfx1250): 3x GATv2 + LN/ELU + policy/value heads.
// Dense GEMMs run on V_WMMA_F32_16X16X32_F16 (f16 inputs, f32 accumulate);
// each wave computes a 16x64 output tile (4 WMMAs share one A fragment).
// Edge softmax/scatter phase uses L2-resident atomics (tables fit in 192MB L2).
// ---------------------------------------------------------------------------

typedef __attribute__((ext_vector_type(16))) _Float16 v16h;
typedef __attribute__((ext_vector_type(8)))  _Float16 v8h;
typedef __attribute__((ext_vector_type(8)))  float    v8f;

#define GN   20000          // nodes
#define GE   320000         // edges (before self loops)
#define GIN  16             // input feature dim
#define GH   4              // heads (layers 1,2)
#define GC   128            // per-head channels
#define GHC  512            // H*C
#define GMLP 128
#define NEG_SLOPE 0.2f

// ------------------------------ utilities ----------------------------------

__device__ __forceinline__ unsigned f2ord(float f) {
    unsigned u = __float_as_uint(f);
    return (u & 0x80000000u) ? ~u : (u | 0x80000000u);
}
__device__ __forceinline__ float ord2f(unsigned u) {
    u = (u & 0x80000000u) ? (u & 0x7fffffffu) : ~u;
    return __uint_as_float(u);
}

__global__ void k_zero_f32(float* __restrict__ p, long long n) {
    long long i = blockIdx.x * (long long)blockDim.x + threadIdx.x;
    if (i < n) p[i] = 0.0f;
}

__global__ void k_reduce_sum(const float* __restrict__ w, int n, float* __restrict__ out) {
    float s = 0.0f;
    for (long long i = blockIdx.x * (long long)blockDim.x + threadIdx.x;
         i < n; i += (long long)gridDim.x * blockDim.x)
        s += w[i];
    #pragma unroll
    for (int off = 16; off; off >>= 1) s += __shfl_xor(s, off, 32);
    if ((threadIdx.x & 31) == 0) atomicAdd(out, s);
}

// f32 -> f16 conversion with zero-padding of the K dimension.
__global__ void k_cvt_pad_f16(const float* __restrict__ src, _Float16* __restrict__ dst,
                              int rows, int Ksrc, int Kdst) {
    long long i = blockIdx.x * (long long)blockDim.x + threadIdx.x;
    long long total = (long long)rows * Kdst;
    if (i >= total) return;
    int r = (int)(i / Kdst);
    int k = (int)(i % Kdst);
    dst[i] = (k < Ksrc) ? (_Float16)src[(size_t)r * Ksrc + k] : (_Float16)0.0f;
}

// ------------------------------ WMMA GEMM ----------------------------------
// out[m,n] = sum_k A[m,k] * W[n,k] + bias[n]   (== A @ W.T + bias)
// A: [M,K] f16 row-major, W: [Nout,K] f16 row-major, out: [M,Nout] f32.
// One wave per 16x64 output tile: 4 accumulators share each A fragment, so
// every s_wait_loadcnt is amortized over 4 back-to-back v_wmma ops.
// Requires M % 16 == 0, K % 32 == 0, Nout % 64 == 0.
__global__ void k_gemm_wmma_bias(const _Float16* __restrict__ A,
                                 const _Float16* __restrict__ W,
                                 const float* __restrict__ bias,
                                 float* __restrict__ out,
                                 int M, int K, int Nout) {
    const int lane = threadIdx.x & 31;
    const int wave = threadIdx.x >> 5;
    const int groupsN = Nout >> 6;                 // 64-wide N groups
    const int tilesM  = M >> 4;
    const int tile = blockIdx.x * (blockDim.x >> 5) + wave;
    if (tile >= tilesM * groupsN) return;          // wave-uniform: EXEC stays all-1s
    const int m0 = (tile / groupsN) << 4;
    const int n0 = (tile % groupsN) << 6;
    const int half = lane >> 4;                    // 0 or 1
    const int l15  = lane & 15;

    // C/D layout: lane -> n = l15 (+16*t), VGPR j -> m = 8*half + j.
    v8f acc[4];
    #pragma unroll
    for (int t = 0; t < 4; ++t) {
        const float bn = bias[n0 + 16 * t + l15];
        #pragma unroll
        for (int j = 0; j < 8; ++j) acc[t][j] = bn;
    }

    const _Float16* arow = A + (size_t)(m0 + l15) * K;   // A fragment row m = l15
    const _Float16* wrow = W + (size_t)(n0 + l15) * K;   // B col n = l15 (+16*t rows)

    for (int kk = 0; kk < K; kk += 32) {
        // A (16x32 f16): elem i -> K = kk + 16*half + i (contiguous, matches B).
        const v8h a_lo = *(const v8h*)(arow + kk + 16 * half);
        const v8h a_hi = *(const v8h*)(arow + kk + 16 * half + 8);
        v16h a;
        #pragma unroll
        for (int i = 0; i < 8; ++i) { a[i] = a_lo[i]; a[i + 8] = a_hi[i]; }

        if (kk + 32 < K) __builtin_prefetch(arow + kk + 32, 0, 1);

        // B (32x16 f16): elem i -> K = kk + 16*half + i (contiguous 32B per lane).
        #pragma unroll
        for (int t = 0; t < 4; ++t) {
            const v16h b = *(const v16h*)(wrow + (size_t)(16 * t) * K + kk + 16 * half);
            acc[t] = __builtin_amdgcn_wmma_f32_16x16x32_f16(
                /*neg_a=*/false, a, /*neg_b=*/false, b,
                /*c_mod=*/(short)0, acc[t], /*reuse_a=*/false, /*reuse_b=*/false);
        }
    }

    #pragma unroll
    for (int t = 0; t < 4; ++t) {
        float* ocol = out + (size_t)n0 + 16 * t + l15;
        #pragma unroll
        for (int j = 0; j < 8; ++j)
            ocol[(size_t)(m0 + 8 * half + j) * Nout] = acc[t][j];
    }
}

// ---------------------------- edge attention -------------------------------
// One wave per (edge, head): alpha_logit = sum_c lrelu(xl[s]+xr[d]+ea*We)*att.
__global__ void k_edge_logits(const float* __restrict__ XL, const float* __restrict__ XR,
                              const int* __restrict__ srcE, const int* __restrict__ dstE,
                              const float* __restrict__ ew, const float* __restrict__ meanSum,
                              const float* __restrict__ We, const float* __restrict__ att,
                              float* __restrict__ logit,
                              int E, int Nn, int Hh, int Cc) {
    const int lane = threadIdx.x & 31;
    const long long gw = (blockIdx.x * (long long)blockDim.x + threadIdx.x) >> 5;
    const long long tot = (long long)(E + Nn) * Hh;
    if (gw >= tot) return;
    const int e = (int)(gw / Hh);
    const int h = (int)(gw % Hh);
    int s, d; float eav;
    if (e < E) { s = srcE[e]; d = dstE[e]; eav = ew[e]; }
    else       { s = d = e - E; eav = meanSum[0] * (1.0f / E); }
    const int HCl = Hh * Cc;
    const float* xlp = XL + (size_t)s * HCl + h * Cc;
    const float* xrp = XR + (size_t)d * HCl + h * Cc;
    const float* wep = We + h * Cc;
    const float* atp = att + h * Cc;
    float sum = 0.0f;
    for (int c = lane; c < Cc; c += 32) {
        float m = xlp[c] + xrp[c] + eav * wep[c];
        m = (m > 0.0f) ? m : m * NEG_SLOPE;
        sum += m * atp[c];
    }
    #pragma unroll
    for (int off = 16; off; off >>= 1) sum += __shfl_xor(sum, off, 32);
    if (lane == 0) logit[gw] = sum;
}

__global__ void k_seg_max(const float* __restrict__ logit, const int* __restrict__ dstE,
                          unsigned* __restrict__ amaxU, int E, int Nn, int Hh) {
    long long i = blockIdx.x * (long long)blockDim.x + threadIdx.x;
    long long tot = (long long)(E + Nn) * Hh;
    if (i >= tot) return;
    int e = (int)(i / Hh), h = (int)(i % Hh);
    int d = (e < E) ? dstE[e] : (e - E);
    atomicMax(&amaxU[(size_t)d * Hh + h], f2ord(logit[i]));
}

__global__ void k_exp_sum(float* __restrict__ logit, const int* __restrict__ dstE,
                          const unsigned* __restrict__ amaxU, float* __restrict__ denom,
                          int E, int Nn, int Hh) {
    long long i = blockIdx.x * (long long)blockDim.x + threadIdx.x;
    long long tot = (long long)(E + Nn) * Hh;
    if (i >= tot) return;
    int e = (int)(i / Hh), h = (int)(i % Hh);
    int d = (e < E) ? dstE[e] : (e - E);
    float a = expf(logit[i] - ord2f(amaxU[(size_t)d * Hh + h]));
    logit[i] = a;
    atomicAdd(&denom[(size_t)d * Hh + h], a);
}

__global__ void k_norm_alpha(float* __restrict__ alpha, const int* __restrict__ dstE,
                             const float* __restrict__ denom, int E, int Nn, int Hh) {
    long long i = blockIdx.x * (long long)blockDim.x + threadIdx.x;
    long long tot = (long long)(E + Nn) * Hh;
    if (i >= tot) return;
    int e = (int)(i / Hh), h = (int)(i % Hh);
    int d = (e < E) ? dstE[e] : (e - E);
    alpha[i] = alpha[i] / (denom[(size_t)d * Hh + h] + 1e-16f);
}

// One block per edge; each thread scatters one float4 chunk of xl[src]*alpha.
__global__ void k_scatter(const float* __restrict__ XL, const float* __restrict__ alpha,
                          const int* __restrict__ srcE, const int* __restrict__ dstE,
                          float* __restrict__ OUT, int E, int Nn, int Hh, int Cc) {
    const int e = blockIdx.x;
    const int t = threadIdx.x;          // chunk index (4 floats)
    int s, d;
    if (e < E) { s = srcE[e]; d = dstE[e]; } else { s = d = e - E; }
    const int h = (t * 4) / Cc;
    const float a = alpha[(size_t)e * Hh + h];
    const int HCl = Hh * Cc;
    const float4 xv = ((const float4*)(XL + (size_t)s * HCl))[t];
    float* op = OUT + (size_t)d * HCl + t * 4;
    atomicAdd(op + 0, xv.x * a);
    atomicAdd(op + 1, xv.y * a);
    atomicAdd(op + 2, xv.z * a);
    atomicAdd(op + 3, xv.w * a);
}

// Wave-per-node LayerNorm((x + bo)) * g + b, optional ELU. In-place safe.
__global__ void k_ln_elu(const float* __restrict__ X, const float* __restrict__ bo,
                         const float* __restrict__ g, const float* __restrict__ b,
                         float* __restrict__ Y, int Nn, int D, int doElu) {
    const int lane = threadIdx.x & 31;
    const int node = (int)((blockIdx.x * (long long)blockDim.x + threadIdx.x) >> 5);
    if (node >= Nn) return;
    const float* xp = X + (size_t)node * D;
    float* yp = Y + (size_t)node * D;
    float s = 0.0f;
    for (int c = lane; c < D; c += 32) s += xp[c] + bo[c];
    #pragma unroll
    for (int off = 16; off; off >>= 1) s += __shfl_xor(s, off, 32);
    const float mu = s / D;
    float v = 0.0f;
    for (int c = lane; c < D; c += 32) { float t = xp[c] + bo[c] - mu; v += t * t; }
    #pragma unroll
    for (int off = 16; off; off >>= 1) v += __shfl_xor(v, off, 32);
    const float inv = rsqrtf(v / D + 1e-5f);
    for (int c = lane; c < D; c += 32) {
        float y = (xp[c] + bo[c] - mu) * inv * g[c] + b[c];
        if (doElu) y = (y > 0.0f) ? y : (expf(y) - 1.0f);
        yp[c] = y;
    }
}

// Final policy/value heads for one node. Single block of 128 threads.
__global__ void k_heads(const float* __restrict__ Hf, const int* __restrict__ nodeIdx,
                        const float* __restrict__ Pw1, const float* __restrict__ Pb1,
                        const float* __restrict__ Pw2, const float* __restrict__ Pb2,
                        const float* __restrict__ Vw1, const float* __restrict__ Vb1,
                        const float* __restrict__ Vw2, const float* __restrict__ Vb2,
                        float* __restrict__ out) {
    __shared__ float z[GC], hp[GMLP], hv[GMLP];
    const int t = threadIdx.x;
    z[t] = Hf[(size_t)nodeIdx[0] * GC + t];
    __syncthreads();
    float sp = Pb1[t], sv = Vb1[t];
    for (int c = 0; c < GC; ++c) { sp += z[c] * Pw1[t * GC + c]; sv += z[c] * Vw1[t * GC + c]; }
    hp[t] = fmaxf(sp, 0.0f);
    hv[t] = fmaxf(sv, 0.0f);
    __syncthreads();
    if (t < 2) { float s = Pb2[t]; for (int c = 0; c < GMLP; ++c) s += hp[c] * Pw2[t * GMLP + c]; out[t] = s; }
    if (t == 2) { float s = Vb2[0]; for (int c = 0; c < GMLP; ++c) s += hv[c] * Vw2[c]; out[2] = s; }
}

// ------------------------------- launcher ----------------------------------

extern "C" void kernel_launch(void* const* d_in, const int* in_sizes, int n_in,
                              void* d_out, int out_size, void* d_ws, size_t ws_size,
                              hipStream_t stream) {
    (void)in_sizes; (void)n_in; (void)out_size; (void)ws_size;

    const float* x    = (const float*)d_in[0];
    const int*   ei   = (const int*)d_in[1];
    const float* ew   = (const float*)d_in[2];
    const int*   nidx = (const int*)d_in[3];
    const float *Wl1 = (const float*)d_in[4],  *bl1 = (const float*)d_in[5];
    const float *Wr1 = (const float*)d_in[6],  *br1 = (const float*)d_in[7];
    const float *We1 = (const float*)d_in[8],  *att1= (const float*)d_in[9],  *bo1 = (const float*)d_in[10];
    const float *Wl2 = (const float*)d_in[11], *bl2 = (const float*)d_in[12];
    const float *Wr2 = (const float*)d_in[13], *br2 = (const float*)d_in[14];
    const float *We2 = (const float*)d_in[15], *att2= (const float*)d_in[16], *bo2 = (const float*)d_in[17];
    const float *Wl3 = (const float*)d_in[18], *bl3 = (const float*)d_in[19];
    const float *Wr3 = (const float*)d_in[20], *br3 = (const float*)d_in[21];
    const float *We3 = (const float*)d_in[22], *att3= (const float*)d_in[23], *bo3 = (const float*)d_in[24];
    const float *g1  = (const float*)d_in[25], *b1  = (const float*)d_in[26];
    const float *g2  = (const float*)d_in[27], *b2  = (const float*)d_in[28];
    const float *g3  = (const float*)d_in[29], *b3  = (const float*)d_in[30];
    const float *Pw1 = (const float*)d_in[31], *Pb1 = (const float*)d_in[32];
    const float *Pw2 = (const float*)d_in[33], *Pb2 = (const float*)d_in[34];
    const float *Vw1 = (const float*)d_in[35], *Vb1 = (const float*)d_in[36];
    const float *Vw2 = (const float*)d_in[37], *Vb2 = (const float*)d_in[38];

    const int* srcE = ei;          // edge_index[0]
    const int* dstE = ei + GE;     // edge_index[1]

    // Carve workspace (256B aligned slices).
    char* base = (char*)d_ws;
    size_t off = 0;
    auto carve = [&](size_t bytes) -> void* {
        off = (off + 255) & ~(size_t)255;
        void* p = base + off;
        off += bytes;
        return p;
    };
    float*     Hbuf = (float*)    carve((size_t)GN * GHC * 4);  // layer io / scatter dst
    _Float16*  A16  = (_Float16*) carve((size_t)GN * GHC * 2);  // f16 activations
    float*     XL   = (float*)    carve((size_t)GN * GHC * 4);
    float*     XR   = (float*)    carve((size_t)GN * GHC * 4);
    _Float16*  W16  = (_Float16*) carve((size_t)GHC * GHC * 2); // f16 weights (reused)
    float*     ALPH = (float*)    carve((size_t)(GE + GN) * GH * 4);
    unsigned*  AMAX = (unsigned*) carve((size_t)GN * GH * 4);
    float*     DEN  = (float*)    carve((size_t)GN * GH * 4);
    float*     MEAN = (float*)    carve(256);

    auto cdiv = [](long long a, long long b) -> int { return (int)((a + b - 1) / b); };

    // Mean edge weight (used as self-loop edge attr).
    k_zero_f32<<<1, 256, 0, stream>>>(MEAN, 1);
    k_reduce_sum<<<256, 256, 0, stream>>>(ew, GE, MEAN);

    auto run_layer = [&](const float* inAct, int Kin, int Ka,
                         const float* Wl, const float* bl,
                         const float* Wr, const float* br,
                         const float* We, const float* att, const float* bo,
                         const float* g, const float* b,
                         int Hh, int Cc, int doElu) {
        const int Nout = Hh * Cc;
        const long long na = (long long)GN * Ka;
        const long long nw = (long long)Nout * Ka;
        const int tiles = (GN / 16) * (Nout / 64);   // 16x64 per wave
        const long long totEH = (long long)(GE + GN) * Hh;

        k_cvt_pad_f16<<<cdiv(na, 256), 256, 0, stream>>>(inAct, A16, GN, Kin, Ka);
        k_cvt_pad_f16<<<cdiv(nw, 256), 256, 0, stream>>>(Wl, W16, Nout, Kin, Ka);
        k_gemm_wmma_bias<<<cdiv(tiles, 8), 256, 0, stream>>>(A16, W16, bl, XL, GN, Ka, Nout);
        k_cvt_pad_f16<<<cdiv(nw, 256), 256, 0, stream>>>(Wr, W16, Nout, Kin, Ka);
        k_gemm_wmma_bias<<<cdiv(tiles, 8), 256, 0, stream>>>(A16, W16, br, XR, GN, Ka, Nout);

        k_zero_f32<<<cdiv((long long)GN * Nout, 256), 256, 0, stream>>>(Hbuf, (long long)GN * Nout);
        k_zero_f32<<<cdiv((long long)GN * Hh, 256), 256, 0, stream>>>((float*)AMAX, (long long)GN * Hh);
        k_zero_f32<<<cdiv((long long)GN * Hh, 256), 256, 0, stream>>>(DEN, (long long)GN * Hh);

        k_edge_logits<<<cdiv(totEH, 8), 256, 0, stream>>>(XL, XR, srcE, dstE, ew, MEAN,
                                                          We, att, ALPH, GE, GN, Hh, Cc);
        k_seg_max<<<cdiv(totEH, 256), 256, 0, stream>>>(ALPH, dstE, AMAX, GE, GN, Hh);
        k_exp_sum<<<cdiv(totEH, 256), 256, 0, stream>>>(ALPH, dstE, AMAX, DEN, GE, GN, Hh);
        k_norm_alpha<<<cdiv(totEH, 256), 256, 0, stream>>>(ALPH, dstE, DEN, GE, GN, Hh);
        k_scatter<<<GE + GN, Nout / 4, 0, stream>>>(XL, ALPH, srcE, dstE, Hbuf, GE, GN, Hh, Cc);
        k_ln_elu<<<cdiv((long long)GN * 32, 256), 256, 0, stream>>>(Hbuf, bo, g, b, Hbuf, GN, Nout, doElu);
    };

    // Layer 1: in=16 (K padded to 32) -> 4x128 concat, ELU(LN)
    run_layer(x, GIN, 32, Wl1, bl1, Wr1, br1, We1, att1, bo1, g1, b1, GH, GC, 1);
    // Layer 2: 512 -> 4x128 concat, ELU(LN)
    run_layer(Hbuf, GHC, GHC, Wl2, bl2, Wr2, br2, We2, att2, bo2, g2, b2, GH, GC, 1);
    // Layer 3: 512 -> 1x128 (mean over 1 head == identity), LN only
    run_layer(Hbuf, GHC, GHC, Wl3, bl3, Wr3, br3, We3, att3, bo3, g3, b3, 1, GC, 0);

    // Policy / value heads for the selected node. d_out = [logit0, logit1, value].
    k_heads<<<1, GMLP, 0, stream>>>(Hbuf, nidx, Pw1, Pb1, Pw2, Pb2,
                                    Vw1, Vb1, Vw2, Vb2, (float*)d_out);
}
